// Encoder_53437983096940
// MI455X (gfx1250) — matrix operand order repaired and checked by reading, not compiled, not run on hardware
//
#include <hip/hip_runtime.h>
#include <hip/hip_bf16.h>

// ---------------------------------------------------------------------------
// 2-layer GRU (PyTorch gate order r,z,n), B=128, T=1024, N=M=256.
//   1) convert X / weights to bf16
//   2) full-GPU WMMA GEMM: gates_x = X @ W_ih^T + b_ih  (stored bf16)
//   3) batch-split scan: 8 WGs (16 batch rows), 8 waves/WG; wave owns 32
//      hidden cols of all 3 gates. W_hh: 6/8 K-blocks staged in LDS once
//      (295 KB), 2/8 streamed from L2. gates_x loads hoisted above the
//      WMMA chain each step. h: fp32 in D-layout VGPRs + bf16 A-tile in LDS.
// ---------------------------------------------------------------------------

typedef __bf16   bf16;
typedef bf16     v8bf  __attribute__((ext_vector_type(8)));
typedef bf16     v16bf __attribute__((ext_vector_type(16)));
typedef float    v8f   __attribute__((ext_vector_type(8)));

#define B_   128
#define T_   1024
#define N_   256   // input features (= K of input GEMM)
#define M_   256   // hidden size
#define G3_  768   // 3*M
#define KBL_ 6     // K-blocks of W_hh staged in LDS (of 8)

// LDS: hA (16x256 bf16) + staged W slots (8 waves * 6 tiles * KBL_ * 512 bf16)
#define HA_BYTES   (16 * M_ * 2)
#define WL_BYTES   (8 * 6 * KBL_ * 512 * 2)
#define SMEM_BYTES (HA_BYTES + WL_BYTES)

static __device__ __forceinline__ v8f zero8() {
  v8f z = {0.f, 0.f, 0.f, 0.f, 0.f, 0.f, 0.f, 0.f};
  return z;
}

static __device__ __forceinline__ v16bf cat16(v8bf lo, v8bf hi) {
  v16bf r;
#pragma unroll
  for (int i = 0; i < 8; ++i) { r[i] = lo[i]; r[8 + i] = hi[i]; }
  return r;
}

// A-matrix (16x32 bf16) loader, row-major source with leading dim ld.
static __device__ __forceinline__ v16bf load_A(const bf16* base, int ld, int k, int lane) {
  const int row = lane & 15;
  const int k16 = (lane >> 4) * 16;
  const bf16* p = base + (long)row * ld + k + k16;
  v8bf lo = *reinterpret_cast<const v8bf*>(p);
  v8bf hi = *reinterpret_cast<const v8bf*>(p + 8);
  return cat16(lo, hi);
}

// B-matrix (32x16 bf16) loader from W row-major [col][K]  (C = A @ W^T).
static __device__ __forceinline__ v16bf load_B(const bf16* w, int colbase, int ld, int k, int lane) {
  const int c   = lane & 15;
  const int k16 = (lane >> 4) * 16;
  const bf16* p = w + (long)(colbase + c) * ld + k + k16;
  v8bf lo = *reinterpret_cast<const v8bf*>(p);
  v8bf hi = *reinterpret_cast<const v8bf*>(p + 8);
  return cat16(lo, hi);
}

static __device__ __forceinline__ v8f wmma_bf16(v16bf a, v16bf b, v8f c) {
  return __builtin_amdgcn_wmma_f32_16x16x32_bf16(false, a, false, b, (short)0, c, false, false);
}

static __device__ __forceinline__ float sigmoidf_(float x) {
  return 1.0f / (1.0f + __expf(-x));
}

// ---------------------------------------------------------------------------
__global__ void cvt_bf16_kernel(const float* __restrict__ in, bf16* __restrict__ out, long n) {
  long i = (long)blockIdx.x * blockDim.x + threadIdx.x;
  long stride = (long)gridDim.x * blockDim.x;
  for (; i < n; i += stride) out[i] = (bf16)in[i];
}

// ---------------------------------------------------------------------------
// gates_x GEMM:  C[rows,768] = A[rows,256] @ W^T + bias, stored bf16.
// grid = (rows/128, 768/64), block = 256. B tiles loaded as a group so the
// 4 WMMAs don't each stall on loadcnt 0.
// ---------------------------------------------------------------------------
__global__ void __launch_bounds__(256)
gru_input_gemm(const bf16* __restrict__ A, const bf16* __restrict__ W,
               const float* __restrict__ bias, bf16* __restrict__ C) {
  const int lane = threadIdx.x & 31;
  const int wave = threadIdx.x >> 5;
  const long r0  = (long)blockIdx.x * 128 + wave * 16;
  const int  c0  = blockIdx.y * 64;

  v8f acc[4];
#pragma unroll
  for (int ct = 0; ct < 4; ++ct) acc[ct] = zero8();

  const bf16* Arow = A + r0 * N_;
#pragma unroll
  for (int kb = 0; kb < N_ / 32; ++kb) {
    const int k = kb * 32;
    v16bf a = load_A(Arow, N_, k, lane);
    v16bf b0 = load_B(W, c0 + 0 * 16, N_, k, lane);
    v16bf b1 = load_B(W, c0 + 1 * 16, N_, k, lane);
    v16bf b2 = load_B(W, c0 + 2 * 16, N_, k, lane);
    v16bf b3 = load_B(W, c0 + 3 * 16, N_, k, lane);
    acc[0] = wmma_bf16(a, b0, acc[0]);
    acc[1] = wmma_bf16(a, b1, acc[1]);
    acc[2] = wmma_bf16(a, b2, acc[2]);
    acc[3] = wmma_bf16(a, b3, acc[3]);
  }

  const int c  = lane & 15;
  const int mh = (lane >> 4) * 8;
#pragma unroll
  for (int ct = 0; ct < 4; ++ct) {
    const int col = c0 + ct * 16 + c;
    const float bv = bias[col];
#pragma unroll
    for (int j = 0; j < 8; ++j) {
      const long row = r0 + mh + j;
      C[row * G3_ + col] = (bf16)(acc[ct][j] + bv);
    }
  }
}

// ---------------------------------------------------------------------------
// Recurrent scan. Block = 16 batch rows, 8 waves; wave owns tiles
// ti = g*2+q  (g = gate r/z/n, q = 0/1) -> cols [32w, 32w+32) per gate.
// W_hh K-blocks 0..KBL_-1 read from LDS (staged once), rest from global.
// ---------------------------------------------------------------------------
template <bool OUT_BF16>
__global__ void __launch_bounds__(256)
gru_scan(const bf16* __restrict__ G, const bf16* __restrict__ Whh,
         const float* __restrict__ bhh, void* __restrict__ outp,
         float* __restrict__ hN) {
  extern __shared__ char smem[];
  bf16* hA = (bf16*)smem;                 // [16][256] bf16 A-layout h tile
  bf16* Wl = (bf16*)(smem + HA_BYTES);    // staged W_hh slots

  const int lane = threadIdx.x & 31;
  const int wave = threadIdx.x >> 5;
  const int r0   = blockIdx.x * 16;
  const int c    = lane & 15;
  const int mh   = (lane >> 4) * 8;
  const int k16  = (lane >> 4) * 16;

  // zero initial hidden state tile
  for (int i = threadIdx.x; i < 16 * M_; i += blockDim.x) hA[i] = (bf16)0.0f;

  // ---- stage this wave's W_hh slice (6 tiles x KBL_ k-blocks) into LDS ----
  // slot(wave,ti,kb): 32 lanes * 16 bf16, lane-private contiguous 32B.
#pragma unroll
  for (int ti = 0; ti < 6; ++ti) {
    const int g = ti >> 1, q = ti & 1;
    const int colbase = g * M_ + (wave * 2 + q) * 16;
    const bf16* gp = Whh + (long)(colbase + c) * M_ + k16;
#pragma unroll
    for (int kb = 0; kb < KBL_; ++kb) {
      v8bf lo = *reinterpret_cast<const v8bf*>(gp + kb * 32);
      v8bf hi = *reinterpret_cast<const v8bf*>(gp + kb * 32 + 8);
      bf16* lp = Wl + (size_t)((wave * 6 + ti) * KBL_ + kb) * 512 + lane * 16;
      *reinterpret_cast<v8bf*>(lp)     = lo;
      *reinterpret_cast<v8bf*>(lp + 8) = hi;
    }
  }

  float hreg[2][8];
#pragma unroll
  for (int q = 0; q < 2; ++q)
#pragma unroll
    for (int j = 0; j < 8; ++j) hreg[q][j] = 0.0f;

  int   colq[2];
  float bR[2], bZ[2], bN[2];
#pragma unroll
  for (int q = 0; q < 2; ++q) {
    colq[q] = wave * 32 + q * 16 + c;
    bR[q] = bhh[colq[q]];
    bZ[q] = bhh[M_ + colq[q]];
    bN[q] = bhh[2 * M_ + colq[q]];
  }
  __syncthreads();

  for (int t = 0; t < T_; ++t) {
    // ---- hoist gates_x loads: overlap global latency with the WMMA chain ----
    float gx[2][3][8];
#pragma unroll
    for (int q = 0; q < 2; ++q) {
#pragma unroll
      for (int j = 0; j < 8; ++j) {
        const long gb = ((long)(r0 + mh + j) * T_ + t) * G3_;
        gx[q][0][j] = (float)G[gb + colq[q]];
        gx[q][1][j] = (float)G[gb + M_ + colq[q]];
        gx[q][2][j] = (float)G[gb + 2 * M_ + colq[q]];
        if (j == 0 && t + 1 < T_) {
          __builtin_prefetch(&G[gb + G3_ + colq[q]], 0, 1);
        }
      }
    }

    // ---- gh = h @ Whh^T : 8 k-blocks x 6 tiles = 48 WMMA/wave ----
    v8f acc[6];
#pragma unroll
    for (int ti = 0; ti < 6; ++ti) acc[ti] = zero8();

#pragma unroll
    for (int kb = 0; kb < M_ / 32; ++kb) {
      const int k = kb * 32;
      v16bf a = load_A(hA, M_, k, lane);
#pragma unroll
      for (int half = 0; half < 2; ++half) {
        v16bf b[3];
#pragma unroll
        for (int i = 0; i < 3; ++i) {
          const int ti = half * 3 + i;
          if (kb < KBL_) {  // from LDS (staged)
            const bf16* lp = Wl + (size_t)((wave * 6 + ti) * KBL_ + kb) * 512 + lane * 16;
            v8bf lo = *reinterpret_cast<const v8bf*>(lp);
            v8bf hi = *reinterpret_cast<const v8bf*>(lp + 8);
            b[i] = cat16(lo, hi);
          } else {          // streamed from L2
            const int g = ti >> 1, q = ti & 1;
            b[i] = load_B(Whh, g * M_ + (wave * 2 + q) * 16, M_, k, lane);
          }
        }
#pragma unroll
        for (int i = 0; i < 3; ++i) {
          acc[half * 3 + i] = wmma_bf16(a, b[i], acc[half * 3 + i]);
        }
      }
    }

    // ---- gate fusion + h update (wave-local, D layout) ----
#pragma unroll
    for (int q = 0; q < 2; ++q) {
#pragma unroll
      for (int j = 0; j < 8; ++j) {
        // acc tile index: ti = g*2 + q
        float r = sigmoidf_(gx[q][0][j] + acc[0 * 2 + q][j] + bR[q]);
        float z = sigmoidf_(gx[q][1][j] + acc[1 * 2 + q][j] + bZ[q]);
        float n = tanhf(gx[q][2][j] + r * (acc[2 * 2 + q][j] + bN[q]));
        float h = (1.0f - z) * n + z * hreg[q][j];
        hreg[q][j] = h;
        const long row = (long)(r0 + mh + j) * T_ + t;
        if (OUT_BF16) {
          ((bf16*)outp)[row * M_ + colq[q]] = (bf16)h;
        } else {
          ((float*)outp)[row * M_ + colq[q]] = h;
        }
      }
    }

    __syncthreads();  // all waves done reading hA this step
#pragma unroll
    for (int q = 0; q < 2; ++q)
#pragma unroll
      for (int j = 0; j < 8; ++j)
        hA[(mh + j) * M_ + colq[q]] = (bf16)hreg[q][j];
    __syncthreads();  // hA updated before next step's WMMA reads
  }

  // final hidden state h_n (fp32)
#pragma unroll
  for (int q = 0; q < 2; ++q)
#pragma unroll
    for (int j = 0; j < 8; ++j)
      hN[(long)(r0 + mh + j) * M_ + colq[q]] = hreg[q][j];
}

// ---------------------------------------------------------------------------
extern "C" void kernel_launch(void* const* d_in, const int* in_sizes, int n_in,
                              void* d_out, int out_size, void* d_ws, size_t ws_size,
                              hipStream_t stream) {
  (void)in_sizes; (void)n_in; (void)out_size; (void)ws_size;

  const float* X      = (const float*)d_in[0];
  const float* w_ih_0 = (const float*)d_in[1];
  const float* w_hh_0 = (const float*)d_in[2];
  const float* b_ih_0 = (const float*)d_in[3];
  const float* b_hh_0 = (const float*)d_in[4];
  const float* w_ih_1 = (const float*)d_in[5];
  const float* w_hh_1 = (const float*)d_in[6];
  const float* b_ih_1 = (const float*)d_in[7];
  const float* b_hh_1 = (const float*)d_in[8];

  const long BT = (long)B_ * T_;
  char* ws = (char*)d_ws;

  bf16* xbf   = (bf16*)(ws);                               // BT*256 (reused as out0 bf16)
  bf16* Gbuf  = (bf16*)(ws + BT * N_ * sizeof(bf16));      // BT*768
  char* wbase = ws + BT * N_ * sizeof(bf16) + BT * G3_ * sizeof(bf16);
  const size_t wbytes = (size_t)G3_ * N_ * sizeof(bf16);
  bf16* wih0 = (bf16*)(wbase + 0 * wbytes);
  bf16* whh0 = (bf16*)(wbase + 1 * wbytes);
  bf16* wih1 = (bf16*)(wbase + 2 * wbytes);
  bf16* whh1 = (bf16*)(wbase + 3 * wbytes);

  float* out1 = (float*)d_out;
  float* hN0  = out1 + BT * M_;
  float* hN1  = hN0 + (long)B_ * M_;

  cvt_bf16_kernel<<<4096, 256, 0, stream>>>(X, xbf, BT * N_);
  cvt_bf16_kernel<<<192, 256, 0, stream>>>(w_ih_0, wih0, (long)G3_ * N_);
  cvt_bf16_kernel<<<192, 256, 0, stream>>>(w_hh_0, whh0, (long)G3_ * M_);
  cvt_bf16_kernel<<<192, 256, 0, stream>>>(w_ih_1, wih1, (long)G3_ * M_);
  cvt_bf16_kernel<<<192, 256, 0, stream>>>(w_hh_1, whh1, (long)G3_ * M_);

  dim3 ggrid((unsigned)(BT / 128), G3_ / 64);

  gru_input_gemm<<<ggrid, 256, 0, stream>>>(xbf, wih0, b_ih_0, Gbuf);
  gru_scan<true><<<B_ / 16, 256, SMEM_BYTES, stream>>>(Gbuf, whh0, b_hh_0, (void*)xbf, hN0);

  gru_input_gemm<<<ggrid, 256, 0, stream>>>(xbf, wih1, b_ih_1, Gbuf);
  gru_scan<false><<<B_ / 16, 256, SMEM_BYTES, stream>>>(Gbuf, whh1, b_hh_1, (void*)out1, hN1);
}
